// RAFT256_55843164783390
// MI455X (gfx1250) — matrix operand order repaired and checked by reading, not compiled, not run on hardware
//
#include <hip/hip_runtime.h>
#include <hip/hip_bf16.h>

typedef __attribute__((ext_vector_type(16))) _Float16 v16h;
typedef __attribute__((ext_vector_type(8)))  _Float16 v8h;
typedef __attribute__((ext_vector_type(8)))  float    v8f;

#define BB   2
#define HH   80
#define WW   80
#define DD   256
#define NPIX (HH * WW)          // 6400

// ---------------------------------------------------------------------------
// Kernel 1: transpose [C=256, P] f32 -> [P, 256] f16 (channel-contiguous
// columns so the main kernel's gathers are 512B-contiguous vector loads).
// Tile 32 channels x 64 positions through LDS; both directions coalesced.
// ---------------------------------------------------------------------------
__global__ __launch_bounds__(256) void k_transpose_cvt(
    const float* __restrict__ in, _Float16* __restrict__ out) {
  __shared__ float tile[32][65];
  const int b  = blockIdx.z;
  const int p0 = blockIdx.x * 64;
  const int c0 = blockIdx.y * 32;
  const float* src = in + (size_t)b * DD * NPIX;
  _Float16*    dst = out + (size_t)b * DD * NPIX;

  #pragma unroll
  for (int e = threadIdx.x; e < 32 * 64; e += 256) {
    int r = e >> 6, col = e & 63;
    tile[r][col] = src[(size_t)(c0 + r) * NPIX + (p0 + col)];
  }
  __syncthreads();
  #pragma unroll
  for (int e = threadIdx.x; e < 64 * 32; e += 256) {
    int pr = e >> 5, cc = e & 31;
    dst[(size_t)(p0 + pr) * DD + (c0 + cc)] = (_Float16)tile[cc][pr];
  }
}

// ---------------------------------------------------------------------------
// Kernel 2: 2x2 mean pool on [B, Hi, Wi, 256] f16 -> [B, Hi/2, Wi/2, 256].
// One block per output position, 256 threads = channels (coalesced).
// ---------------------------------------------------------------------------
__global__ __launch_bounds__(256) void k_pool2(
    const _Float16* __restrict__ in, _Float16* __restrict__ out,
    int Wi, int Ho, int Wo) {
  const int idx = blockIdx.x;           // b*Ho*Wo + oy*Wo + ox
  const int ox = idx % Wo;
  const int t  = idx / Wo;
  const int oy = t % Ho;
  const int b  = t / Ho;
  const int c  = threadIdx.x;
  const size_t row = (size_t)Wi * DD;
  const size_t ib = (((size_t)b * (2 * Ho) + 2 * oy) * Wi + 2 * ox) * DD + c;
  float v = (float)in[ib] + (float)in[ib + DD] +
            (float)in[ib + row] + (float)in[ib + row + DD];
  out[(size_t)idx * DD + c] = (_Float16)(0.25f * v);
}

// ---------------------------------------------------------------------------
// Main kernel: one block per pixel (12800 blocks), 128 threads = 4 waves,
// wave w handles pyramid level w.
// Per level: D[jy][jx] = dot(f1_col, f2_lvl[y0-4+jy, x0-4+jx]) for a 10x10
// window (zero outside), computed as 7 WMMA M-tiles of 16 positions with
// K=256 split into 8 chunks of v_wmma_f32_16x16x32_f16. B operand = f1
// chunk replicated across the 16 N columns (built once per wave).
// Loads are unconditional (clamped to column 0 when OOB -> L0-resident);
// invalid rows are zeroed when the C fragment is written to LDS, using
// wave-uniform scalar validity. Then out[s=a*9+b9] = 4-tap blend of D with
// the shared fractional weights, * 1/sqrt(256).
// ---------------------------------------------------------------------------
__global__ __launch_bounds__(128) void k_corr_sample(
    const _Float16* __restrict__ f1T,                 // [B, 6400, 256]
    const _Float16* __restrict__ l0,                  // [B, 6400, 256]
    const _Float16* __restrict__ l1,                  // [B, 1600, 256]
    const _Float16* __restrict__ l2,                  // [B,  400, 256]
    const _Float16* __restrict__ l3,                  // [B,  100, 256]
    const float*    __restrict__ coords,              // [B, 2, 80, 80]
    float*          __restrict__ out) {               // [B, 324, 80, 80]
  __shared__ _Float16 f1s[DD];
  __shared__ float    Dls[4][112];

  const int pix  = blockIdx.x;
  const int b    = pix / NPIX;
  const int pos  = pix - b * NPIX;
  const int lane = threadIdx.x & 31;
  const int wave = threadIdx.x >> 5;                  // = pyramid level

  // stage f1 column (already f16) into LDS: 128 threads x 4B
  ((unsigned int*)f1s)[threadIdx.x] =
      ((const unsigned int*)(f1T + (size_t)pix * DD))[threadIdx.x];
  __syncthreads();

  // Build replicated-f1 B fragments: chunk k covers channels [k*32, k*32+32).
  // B uses contiguous K order: lanes 0-15 hold K=0..15, lanes 16-31 hold K=16..31.
  v16h bfrag[8];
  {
    const int boff = (lane < 16) ? 0 : 16;
    #pragma unroll
    for (int k = 0; k < 8; ++k)
      bfrag[k] = *(const v16h*)(f1s + k * 32 + boff);
  }

  const float cx  = coords[((size_t)b * 2 + 0) * NPIX + pos];
  const float cy  = coords[((size_t)b * 2 + 1) * NPIX + pos];
  const float sc  = 1.0f / (float)(1 << wave);
  const float cxl = cx * sc, cyl = cy * sc;
  const float x0f = floorf(cxl), y0f = floorf(cyl);
  const float fx  = cxl - x0f,   fy  = cyl - y0f;
  const int   ix0 = (int)x0f,    iy0 = (int)y0f;
  const int   Wl  = WW >> wave,  Hl  = HH >> wave;

  const _Float16* lvl = (wave == 0) ? l0 : (wave == 1) ? l1
                        : (wave == 2) ? l2 : l3;
  const _Float16* lbase = lvl + (size_t)b * Hl * Wl * DD;

  // A-fragment addressing (contiguous K order, matching B):
  // lanes 0-15 : elems 0..15 = K+0..15
  // lanes 16-31: elems 0..15 = K+16..31
  const int m     = lane & 15;
  const int koff0 = (lane < 16) ? 0 : 16;

  for (int t7 = 0; t7 < 7; ++t7) {
    const int p  = t7 * 16 + m;
    const int jy = p / 10, jx = p - jy * 10;
    const int gx = ix0 - 4 + jx;
    const int gy = iy0 - 4 + jy;
    const bool inb = (gx >= 0) && (gx < Wl) && (gy >= 0) && (gy < Hl);
    const _Float16* colp = lbase + (size_t)(inb ? (gy * Wl + gx) : 0) * DD;

    v8f c = {0.f, 0.f, 0.f, 0.f, 0.f, 0.f, 0.f, 0.f};
    #pragma unroll
    for (int k = 0; k < 8; ++k) {
      v8h a0 = *(const v8h*)(colp + k * 32 + koff0);       // b128
      v8h a1 = *(const v8h*)(colp + k * 32 + koff0 + 8);   // b128
      v16h a;
      #pragma unroll
      for (int i = 0; i < 8; ++i) { a[i] = a0[i]; a[i + 8] = a1[i]; }
      c = __builtin_amdgcn_wmma_f32_16x16x32_f16(
          /*neg_a=*/false, a, /*neg_b=*/false, bfrag[k],
          /*c_mod=*/(short)0, c, /*reuse_a=*/false, /*reuse_b=*/false);
    }

    // C/D f32 16x16 layout: VGPR r = row r (lanes 0-15) / row 8+r (16-31).
    // Column N is irrelevant (B replicated) -> lanes 0 and 16 hold the dots.
    // Zero invalid rows here (validity is uniform given the row index).
    if (lane == 0 || lane == 16) {
      const int rowbase = t7 * 16 + ((lane == 16) ? 8 : 0);
      #pragma unroll
      for (int r = 0; r < 8; ++r) {
        const int pr  = rowbase + r;
        const int rjy = pr / 10, rjx = pr - rjy * 10;
        const int rgx = ix0 - 4 + rjx;
        const int rgy = iy0 - 4 + rjy;
        const bool ok = (pr < 100) && (rgx >= 0) && (rgx < Wl) &&
                        (rgy >= 0) && (rgy < Hl);
        Dls[wave][pr] = ok ? c[r] : 0.f;
      }
    }
  }

  // Blend: sample s = a9*9 + b9 -> x offset (a9-4), y offset (b9-4);
  // corners D[b9..b9+1][a9..a9+1] with shared fractional weights.
  const float w00 = (1.f - fx) * (1.f - fy);
  const float w10 = fx * (1.f - fy);
  const float w01 = (1.f - fx) * fy;
  const float w11 = fx * fy;
  const float inv_sqrtD = 0.0625f;      // 1/sqrt(256)
  const float* Dw = Dls[wave];
  for (int s = lane; s < 81; s += 32) {
    const int a9 = s / 9, b9 = s - a9 * 9;
    float v = w00 * Dw[b9 * 10 + a9]       + w10 * Dw[b9 * 10 + a9 + 1] +
              w01 * Dw[(b9 + 1) * 10 + a9] + w11 * Dw[(b9 + 1) * 10 + a9 + 1];
    out[((size_t)b * 324 + wave * 81 + s) * NPIX + pos] = v * inv_sqrtD;
  }
}

// ---------------------------------------------------------------------------
extern "C" void kernel_launch(void* const* d_in, const int* in_sizes, int n_in,
                              void* d_out, int out_size, void* d_ws,
                              size_t ws_size, hipStream_t stream) {
  const float* fmap1  = (const float*)d_in[0];
  const float* fmap2  = (const float*)d_in[1];
  const float* coords = (const float*)d_in[2];
  float* out = (float*)d_out;

  // workspace layout (f16), ~14.6 MB total
  char* ws = (char*)d_ws;
  const size_t n_full = (size_t)BB * NPIX * DD;   // 3,276,800 halfs
  _Float16* f1T = (_Float16*)ws;
  _Float16* l0  = f1T + n_full;
  _Float16* l1  = l0 + n_full;                    // [B,40,40,256]
  _Float16* l2  = l1 + (size_t)BB * 1600 * DD;    // [B,20,20,256]
  _Float16* l3  = l2 + (size_t)BB * 400 * DD;     // [B,10,10,256]

  dim3 tgrid(NPIX / 64, DD / 32, BB);
  k_transpose_cvt<<<tgrid, 256, 0, stream>>>(fmap1, f1T);
  k_transpose_cvt<<<tgrid, 256, 0, stream>>>(fmap2, l0);

  k_pool2<<<BB * 40 * 40, 256, 0, stream>>>(l0, l1, 80, 40, 40);
  k_pool2<<<BB * 20 * 20, 256, 0, stream>>>(l1, l2, 40, 20, 20);
  k_pool2<<<BB * 10 * 10, 256, 0, stream>>>(l2, l3, 20, 10, 10);

  k_corr_sample<<<BB * NPIX, 128, 0, stream>>>(f1T, l0, l1, l2, l3, coords,
                                               out);
}
